// RGCs_FCL_72370198937665
// MI455X (gfx1250) — matrix operand order repaired and checked by reading, not compiled, not run on hardware
//
#include <hip/hip_runtime.h>

// RGC-net on MI455X (gfx1250).
// - bf16 adjacency materialized once (134 MB -> L2-resident, 192 MB L2)
// - all GEMMs via v_wmma_f32_16x16x32_bf16, fp32 accumulation
// - phase-1 B operand (x) staged in LDS per 256-column chunk, double-buffered,
//   via global_load_async_to_lds_b128 (builtin confirmed present by probe)
// - M=32 per wave: two adj row-tiles share each B fragment
// Workspace requirement ~143 MB.

#define NN 2048
#define BB 16
#define CC 64

typedef __attribute__((ext_vector_type(16))) __bf16 v16bf;
typedef __attribute__((ext_vector_type(8)))  float  v8f;
typedef int gv4i __attribute__((vector_size(16)));   // matches builtin proto

union Frag { unsigned int u[8]; v16bf v; };

#if defined(__has_builtin)
#if __has_builtin(__builtin_amdgcn_global_load_async_to_lds_b128) && \
    __has_builtin(__builtin_amdgcn_s_wait_asynccnt)
#define HAVE_ASYNC_LDS 1
#endif
#endif

__device__ __forceinline__ unsigned short f2bf(float f) {
  unsigned int u = __float_as_uint(f);
  u += 0x7FFFu + ((u >> 16) & 1u);          // round-to-nearest-even
  return (unsigned short)(u >> 16);
}
__device__ __forceinline__ float bf2f(unsigned short h) {
  return __uint_as_float(((unsigned int)h) << 16);
}

// ---------------------------------------------------------------- batchnorm
__global__ void bn_kernel(const float* __restrict__ e,
                          const float* __restrict__ phi,
                          const float* __restrict__ eta,
                          unsigned short* __restrict__ x0) {
  int b = blockIdx.x;
  int t = threadIdx.x;                       // 256 threads
  __shared__ float red[256];
  const float* srcs[3] = { e + b * NN, eta + b * NN, phi + b * NN };
  for (int ch = 0; ch < 3; ++ch) {
    const float* s = srcs[ch];
    float sum = 0.f, sq = 0.f;
    for (int n = t; n < NN; n += 256) { float v = s[n]; sum += v; sq += v * v; }
    red[t] = sum; __syncthreads();
    for (int o = 128; o > 0; o >>= 1) { if (t < o) red[t] += red[t + o]; __syncthreads(); }
    float mean = red[0] / NN; __syncthreads();
    red[t] = sq; __syncthreads();
    for (int o = 128; o > 0; o >>= 1) { if (t < o) red[t] += red[t + o]; __syncthreads(); }
    float var = red[0] / NN - mean * mean; __syncthreads();
    float sd = sqrtf(fmaxf(var, 0.f)) + 1e-5f;
    unsigned short* dst = x0 + ((size_t)b * CC + ch) * NN;
    for (int n = t; n < NN; n += 256) dst[n] = f2bf((s[n] - mean) / sd);
  }
  unsigned short* dst = x0 + (size_t)b * CC * NN;
  for (int idx = 3 * NN + t; idx < CC * NN; idx += 256) dst[idx] = 0;
}

// ---------------------------------------------------------- weight packing
__global__ void wprep_kernel(const float* __restrict__ W0, const float* __restrict__ Wr0,
                             const float* __restrict__ W1, const float* __restrict__ Wr1,
                             const float* __restrict__ W2, const float* __restrict__ Wr2,
                             const float* __restrict__ W3, const float* __restrict__ Wr3,
                             unsigned short* __restrict__ Wp,
                             unsigned short* __restrict__ Wrp) {
  int t = threadIdx.x + blockIdx.x * blockDim.x;
  int stride = gridDim.x * blockDim.x;
  const float* Ws[4]  = { W0, W1, W2, W3 };
  const float* Wrs[4] = { Wr0, Wr1, Wr2, Wr3 };
  for (int idx = t; idx < 4 * 64 * 128; idx += stride) {
    int l = idx / (64 * 128); int r = idx % (64 * 128);
    int o = r / 128; int c = r % 128;
    int cin = (l == 0) ? 3 : 64;
    float v = 0.f;
    if (c < 64) { if (c < cin) v = Ws[l][o * 2 * cin + c]; }
    else        { int cc = c - 64; if (cc < cin) v = Ws[l][o * 2 * cin + cin + cc]; }
    Wp[idx] = f2bf(v);
  }
  for (int idx = t; idx < 4 * 64 * 64; idx += stride) {
    int l = idx / (64 * 64); int r = idx % (64 * 64);
    int o = r / 64; int c = r % 64;
    int cin = (l == 0) ? 3 : 64;
    float v = (c < cin) ? Wrs[l][o * cin + c] : 0.f;
    Wrp[idx] = f2bf(v);
  }
}

// -------------------------------------------------------------- adjacency
__global__ void adj_kernel(const float* __restrict__ phi,
                           const float* __restrict__ eta,
                           unsigned short* __restrict__ adj) {
  int blk = blockIdx.x;                      // b*2048 + i
  int b = blk >> 11, i = blk & 2047;
  int t = threadIdx.x;                       // 256 threads, 8 cols each
  float pi = phi[b * NN + i], ei = eta[b * NN + i];
  const float* pj = phi + b * NN;
  const float* ej = eta + b * NN;
  union { unsigned short s[8]; uint4 q; } out;
  int j0 = t * 8;
#pragma unroll
  for (int k = 0; k < 8; ++k) {
    float dp = pi - pj[j0 + k], de = ei - ej[j0 + k];
    out.s[k] = f2bf(__expf(-(dp * dp + de * de)));
  }
  *(uint4*)(adj + (size_t)blk * NN + j0) = out.q;
}

// ----------------------------------------------------------- fused layer
// Block: 128 threads = 4 waves, 128 nodes (32 per wave: two 16-row tiles).
// Phase 1: s = adj @ x^T, x chunks (64c x 256j, 32KB) staged in LDS,
//          double-buffered async; A frags streamed from global adj.
// Phase 2: out = relu(W.[x;s]+b) + Wr.x + br, all WMMA.
__global__ __launch_bounds__(128)
void layer_kernel(const unsigned short* __restrict__ adj,
                  const unsigned short* __restrict__ xin,
                  const unsigned short* __restrict__ Wp,
                  const unsigned short* __restrict__ Wrp,
                  const float* __restrict__ bvec,
                  const float* __restrict__ brvec,
                  unsigned short* __restrict__ xout) {
  __shared__ unsigned short xs[2][64 * 256];     // [buf][c][jj]  2 x 32 KB
  __shared__ unsigned short xi[128 * 64];        // [i_local][c]  16 KB
  __shared__ unsigned short sl[4][32 * 64];      // per-wave s tile [i][c] 16 KB

  int b    = blockIdx.y;
  int iblk = blockIdx.x * 128;
  int t = threadIdx.x, wave = t >> 5, lane = t & 31;
  int g = lane >> 4, ln = lane & 15;

  const unsigned short* xb = xin + (size_t)b * CC * NN;

  // Stage the block's 128-node x tile transposed as [i][c] (phase-2 B frags).
  for (int idx = t; idx < 128 * 64; idx += 128) {
    int c = idx >> 7, ii = idx & 127;
    xi[ii * 64 + c] = xb[c * NN + iblk + ii];
  }

  // Cooperative stage of one 256-column x chunk into LDS buffer.
  auto stage = [&](int jc, unsigned short* dst) {
#pragma unroll
    for (int r = 0; r < 16; ++r) {
      int q = t + r * 128;                    // 2048 16-byte chunks
      int c = q >> 5, part = q & 31;
      const unsigned short* src = xb + c * NN + jc * 256 + part * 8;
      unsigned short* d = dst + c * 256 + part * 8;
#ifdef HAVE_ASYNC_LDS
      __builtin_amdgcn_global_load_async_to_lds_b128((gv4i*)src, (gv4i*)d, 0, 0);
#else
      *(uint4*)d = *(const uint4*)src;
#endif
    }
  };

  int ibase = iblk + wave * 32;
  const unsigned short* arow0 = adj + ((size_t)b * NN + ibase + ln) * NN;
  const unsigned short* arow1 = arow0 + (size_t)16 * NN;

  // ---- Phase 1: fp32 accumulate over K=2048 ----
  v8f acc[2][4];
#pragma unroll
  for (int it = 0; it < 2; ++it)
#pragma unroll
    for (int ct = 0; ct < 4; ++ct) acc[it][ct] = (v8f)0.f;

  stage(0, xs[0]);
  for (int jc = 0; jc < 8; ++jc) {
    unsigned short* cur = xs[jc & 1];
    if (jc + 1 < 8) {
      stage(jc + 1, xs[(jc + 1) & 1]);
#ifdef HAVE_ASYNC_LDS
      __builtin_amdgcn_s_wait_asynccnt(16);   // async done in-order: chunk jc resident
#endif
    } else {
#ifdef HAVE_ASYNC_LDS
      __builtin_amdgcn_s_wait_asynccnt(0);
#endif
    }
    __syncthreads();

    // prefetch next adj chunk for both row tiles
    __builtin_prefetch((const void*)(arow0 + jc * 256 + 256), 0, 0);
    __builtin_prefetch((const void*)(arow1 + jc * 256 + 256), 0, 0);

    for (int jj = 0; jj < 256; jj += 32) {
      Frag a0, a1;
#pragma unroll
      for (int v = 0; v < 8; ++v) {
        int k = 2 * v + ((v >= 4) ? 8 : 0) + 8 * g;
        a0.u[v] = *(const unsigned int*)(arow0 + jc * 256 + jj + k);
        a1.u[v] = *(const unsigned int*)(arow1 + jc * 256 + jj + k);
      }
#pragma unroll
      for (int ct = 0; ct < 4; ++ct) {
        Frag bf;
        const unsigned short* xrow = cur + (ct * 16 + ln) * 256 + jj;
#pragma unroll
        for (int v = 0; v < 8; ++v)
          bf.u[v] = *(const unsigned int*)(xrow + 2 * v + ((v >= 4) ? 8 : 0) + 8 * g);
        acc[0][ct] = __builtin_amdgcn_wmma_f32_16x16x32_bf16(
            false, a0.v, false, bf.v, (short)0, acc[0][ct], false, false);
        acc[1][ct] = __builtin_amdgcn_wmma_f32_16x16x32_bf16(
            false, a1.v, false, bf.v, (short)0, acc[1][ct], false, false);
      }
    }
    __syncthreads();
  }

  // Spill s (D layout: N=c=ln, M=i=v+8g) to LDS as [i][c] bf16.
  unsigned short* sw = &sl[wave][0];
#pragma unroll
  for (int it = 0; it < 2; ++it)
#pragma unroll
    for (int ct = 0; ct < 4; ++ct)
#pragma unroll
      for (int v = 0; v < 8; ++v)
        sw[(it * 16 + v + 8 * g) * 64 + ct * 16 + ln] = f2bf(acc[it][ct][v]);

  // ---- Phase 2: out/res small GEMMs, all WMMA ----
#pragma unroll
  for (int it = 0; it < 2; ++it) {
    const unsigned short* xrowi = &xi[(wave * 32 + it * 16 + ln) * 64];
    const unsigned short* srowi = &sl[wave][(it * 16 + ln) * 64];
    Frag bx[2], bs[2];
#pragma unroll
    for (int ks = 0; ks < 2; ++ks)
#pragma unroll
      for (int v = 0; v < 8; ++v) {
        int kB = ks * 32 + 2 * v + ((v >= 4) ? 8 : 0) + 8 * g;
        bx[ks].u[v] = *(const unsigned int*)(xrowi + kB);
        bs[ks].u[v] = *(const unsigned int*)(srowi + kB);
      }
#pragma unroll
    for (int ot = 0; ot < 4; ++ot) {
      v8f accO = (v8f)0.f, accR = (v8f)0.f;
      int o = ot * 16 + ln;                       // A-frag M index
      const unsigned short* wrow  = Wp  + o * 128;
      const unsigned short* wrrow = Wrp + o * 64;
#pragma unroll
      for (int ks = 0; ks < 2; ++ks) {
        Frag aw, ar, as2;
#pragma unroll
        for (int v = 0; v < 8; ++v) {
          int kA = ks * 32 + 2 * v + ((v >= 4) ? 8 : 0) + 8 * g;
          aw.u[v]  = *(const unsigned int*)(wrow + kA);
          ar.u[v]  = *(const unsigned int*)(wrrow + kA);
          as2.u[v] = *(const unsigned int*)(wrow + 64 + kA);
        }
        accO = __builtin_amdgcn_wmma_f32_16x16x32_bf16(
            false, aw.v, false, bx[ks].v, (short)0, accO, false, false);
        accO = __builtin_amdgcn_wmma_f32_16x16x32_bf16(
            false, as2.v, false, bs[ks].v, (short)0, accO, false, false);
        accR = __builtin_amdgcn_wmma_f32_16x16x32_bf16(
            false, ar.v, false, bx[ks].v, (short)0, accR, false, false);
      }
#pragma unroll
      for (int v = 0; v < 8; ++v) {
        int ov = ot * 16 + v + 8 * g;
        float y = fmaxf(accO[v] + bvec[ov], 0.f) + accR[v] + brvec[ov];
        xout[(size_t)b * CC * NN + (size_t)ov * NN + ibase + it * 16 + ln] = f2bf(y);
      }
    }
  }
}

// ----------------------------------------------------------- pool + FC
__global__ void pool_kernel(const unsigned short* __restrict__ x,
                            const float* __restrict__ fcW,
                            const float* __restrict__ fcb,
                            float* __restrict__ out) {
  int b = blockIdx.x;
  int c = threadIdx.x;                       // 64 threads = channels
  __shared__ float red[64];
  const unsigned short* row = x + ((size_t)b * CC + c) * NN;
  float s = 0.f;
  for (int n = 0; n < NN; n += 2) {
    unsigned int u = *(const unsigned int*)(row + n);
    s += bf2f((unsigned short)(u & 0xFFFF)) + bf2f((unsigned short)(u >> 16));
  }
  red[c] = (s / NN) * fcW[c];
  __syncthreads();
  for (int o = 32; o > 0; o >>= 1) { if (c < o) red[c] += red[c + o]; __syncthreads(); }
  if (c == 0) out[b] = red[0] + fcb[0];
}

extern "C" void kernel_launch(void* const* d_in, const int* in_sizes, int n_in,
                              void* d_out, int out_size, void* d_ws, size_t ws_size,
                              hipStream_t stream) {
  const float* e   = (const float*)d_in[0];
  const float* phi = (const float*)d_in[1];
  const float* eta = (const float*)d_in[2];
  const float *W[4], *bv[4], *Wr[4], *brv[4];
  for (int l = 0; l < 4; ++l) {
    W[l]   = (const float*)d_in[3 + 4 * l];
    bv[l]  = (const float*)d_in[4 + 4 * l];
    Wr[l]  = (const float*)d_in[5 + 4 * l];
    brv[l] = (const float*)d_in[6 + 4 * l];
  }
  const float* fcW = (const float*)d_in[19];
  const float* fcb = (const float*)d_in[20];

  char* ws = (char*)d_ws;
  const size_t ADJ = (size_t)BB * NN * NN * 2;   // 134 MB bf16 adjacency
  const size_t XS  = (size_t)BB * CC * NN * 2;   // 4 MB per x buffer
  unsigned short* adj = (unsigned short*)ws;
  unsigned short* xA  = (unsigned short*)(ws + ADJ);
  unsigned short* xB  = (unsigned short*)(ws + ADJ + XS);
  unsigned short* Wp  = (unsigned short*)(ws + ADJ + 2 * XS);
  unsigned short* Wrp = (unsigned short*)(ws + ADJ + 2 * XS + 4 * 64 * 128 * 2);

  bn_kernel<<<BB, 256, 0, stream>>>(e, phi, eta, xA);
  wprep_kernel<<<32, 256, 0, stream>>>(W[0], Wr[0], W[1], Wr[1], W[2], Wr[2],
                                       W[3], Wr[3], Wp, Wrp);
  adj_kernel<<<BB * NN, 256, 0, stream>>>(phi, eta, adj);

  unsigned short* xs[2] = { xA, xB };
  for (int l = 0; l < 4; ++l) {
    layer_kernel<<<dim3(16, 16), 128, 0, stream>>>(
        adj, xs[l & 1], Wp + l * 64 * 128, Wrp + l * 64 * 64,
        bv[l], brv[l], xs[(l + 1) & 1]);
  }
  pool_kernel<<<BB, 64, 0, stream>>>(xs[0], fcW, fcb, (float*)d_out);
}
